// ContinuousFilterConvolution_87703232184539
// MI455X (gfx1250) — matrix operand order repaired and checked by reading, not compiled, not run on hardware
//
#include <hip/hip_runtime.h>

// ContinuousFilterConvolution fused kernel for gfx1250 (MI455X).
// Shapes: B=32, N=512, K=64 (neighbors), G=64 (rbf), F=128.
// Strategy: one workgroup per (b,n); both GEMMs on v_wmma_f32_16x16x32_bf16
// with fp32 accumulation; everything fused so the [B,N,K,F] intermediates
// (2 x 536 MB) never touch HBM. HBM traffic ~290 MB -> ~12us floor @23.3TB/s.

typedef __attribute__((ext_vector_type(16))) __bf16 v16bf;
typedef __attribute__((ext_vector_type(8)))  float  v8f;

#define BDIM 32
#define NDIM 512
#define KDIM 64
#define GDIM 64
#define FDIM 128

// Pack a [32*nks x 128] f32 weight matrix into per-lane WMMA B-fragment order:
// dst[((ks*8 + nt)*32 + lane)*16 + e] = bf16(W[ks*32 + (lane/16)*8 + (e&7) + (e/8)*16][nt*16 + lane%16])
// This matches the CDNA5 32x16 bf16 B-matrix VGPR layout (lanes 0-15: K=0-7,16-23,
// lanes 16-31: K=8-15,24-31, two K values packed per dword).
__global__ void pack_w_bf16(const float* __restrict__ W, __bf16* __restrict__ dst, int nks) {
  int t = blockIdx.x * blockDim.x + threadIdx.x;
  int total = nks * 8 * 32 * 16;
  if (t >= total) return;
  int e    = t & 15;
  int lane = (t >> 4) & 31;
  int nt   = (t >> 9) & 7;
  int ks   = t >> 12;
  int hi = lane >> 4, l = lane & 15;
  int row = ks * 32 + hi * 8 + (e & 7) + ((e >> 3) << 4);
  int col = nt * 16 + l;
  dst[t] = (__bf16)W[row * FDIM + col];   // native v_cvt bf16 path
}

__global__ __launch_bounds__(256)
void cfconv_fused(const float* __restrict__ features,   // [B,N,F]
                  const float* __restrict__ rbf,        // [B,N,K,G]
                  const int*   __restrict__ nbr,        // [B,N,K]
                  const float* __restrict__ mask,       // [B,N,K]
                  const float* __restrict__ b1,         // [F]
                  const float* __restrict__ b2,         // [F]
                  const __bf16* __restrict__ w1p,       // packed [2][8][32][16]
                  const __bf16* __restrict__ w2p,       // packed [4][8][32][16]
                  float* __restrict__ out)              // [B,N,F]
{
  // Row padding (+8 bf16 = 16B) rotates LDS banks by 4 dwords/row so the
  // 16-lane b128 A-fragment reads are conflict-free.
  __shared__ __align__(16) __bf16 sRbf[KDIM][GDIM + 8];   //  9.0 KB
  __shared__ __align__(16) __bf16 sH[KDIM][FDIM + 8];     // 17.0 KB
  __shared__ float sMask[KDIM];
  __shared__ int   sNbr[KDIM];
  __shared__ float sOut[FDIM];

  const int tid = threadIdx.x;
  const int bn  = blockIdx.x;
  const int b   = bn >> 9;                       // / NDIM
  const size_t rbfBase = (size_t)bn * (KDIM * GDIM);
  const size_t nkBase  = (size_t)bn * KDIM;

  // Stage rbf tile: float4 (b128) global loads -> packed 4xbf16 (b64) LDS
  // stores. 4096 elems / (256 thr * 4) = 4 iterations.
  for (int i = tid * 4; i < KDIM * GDIM; i += 256 * 4) {
    const float4 v = *(const float4*)(rbf + rbfBase + i);
    const int r = i >> 6, c = i & 63;           // 4 elems stay in one row
    union { __bf16 h[4]; uint2 u; } pk;
    pk.h[0] = (__bf16)v.x; pk.h[1] = (__bf16)v.y;
    pk.h[2] = (__bf16)v.z; pk.h[3] = (__bf16)v.w;
    *(uint2*)&sRbf[r][c] = pk.u;
  }
  if (tid < KDIM) {
    sMask[tid] = mask[nkBase + tid];
    sNbr[tid]  = nbr[nkBase + tid];
  }
  if (tid < FDIM) sOut[tid] = 0.0f;
  __syncthreads();

  const int wave = tid >> 5;
  const int lane = tid & 31;
  const int l    = lane & 15;
  const int hi   = lane >> 4;
  const int mt   = wave >> 1;          // 4 m-tiles, 2 waves each
  const int ntB  = (wave & 1) * 4;     // each wave owns 4 n-tiles
  const int m0   = mt * 16;
  const int rowA = m0 + l;             // A-fragment row (same for lanes l and l+16)

  const v8f vzero = {0.f, 0.f, 0.f, 0.f, 0.f, 0.f, 0.f, 0.f};

  // ---- GEMM1: H = shifted_softplus(rbf @ W1 + b1), M=64 N=128 K=64 ----
  v8f acc[4];
  #pragma unroll
  for (int j = 0; j < 4; ++j) acc[j] = vzero;

  #pragma unroll
  for (int ks = 0; ks < GDIM / 32; ++ks) {
    union { v16bf v; uint4 q[2]; } a;
    a.q[0] = *(const uint4*)&sRbf[rowA][ks * 32 + hi * 8];
    a.q[1] = *(const uint4*)&sRbf[rowA][ks * 32 + 16 + hi * 8];
    #pragma unroll
    for (int j = 0; j < 4; ++j) {
      const int nt = ntB + j;
      union { v16bf v; uint4 q[2]; } bb;
      const uint4* p = (const uint4*)(w1p + (((ks * 8 + nt) * 32 + lane) << 4));
      bb.q[0] = p[0]; bb.q[1] = p[1];
      acc[j] = __builtin_amdgcn_wmma_f32_16x16x32_bf16(false, a.v, false, bb.v,
                                                       (short)0, acc[j], false, false);
    }
  }

  // Epilogue 1: +b1, shifted softplus, store bf16 H to LDS.
  #pragma unroll
  for (int j = 0; j < 4; ++j) {
    const int col = (ntB + j) * 16 + l;
    const float bias = b1[col];
    #pragma unroll
    for (int i = 0; i < 8; ++i) {
      float x  = acc[j][i] + bias;
      float sp = fmaxf(x, 0.0f) + log1pf(expf(-fabsf(x))) - 0.69314718056f;
      sH[m0 + hi * 8 + i][col] = (__bf16)sp;
    }
  }
  __syncthreads();

  // ---- GEMM2: filt = H @ W2 + b2, M=64 N=128 K=128 ----
  v8f acc2[4];
  #pragma unroll
  for (int j = 0; j < 4; ++j) acc2[j] = vzero;

  #pragma unroll
  for (int ks = 0; ks < FDIM / 32; ++ks) {
    union { v16bf v; uint4 q[2]; } a;
    a.q[0] = *(const uint4*)&sH[rowA][ks * 32 + hi * 8];
    a.q[1] = *(const uint4*)&sH[rowA][ks * 32 + 16 + hi * 8];
    #pragma unroll
    for (int j = 0; j < 4; ++j) {
      const int nt = ntB + j;
      union { v16bf v; uint4 q[2]; } bb;
      const uint4* p = (const uint4*)(w2p + (((ks * 8 + nt) * 32 + lane) << 4));
      bb.q[0] = p[0]; bb.q[1] = p[1];
      acc2[j] = __builtin_amdgcn_wmma_f32_16x16x32_bf16(false, a.v, false, bb.v,
                                                        (short)0, acc2[j], false, false);
    }
  }

  // ---- Epilogue 2: gather neighbor features (L2-resident), mask, K-reduce ----
  const float* featB = features + (size_t)b * (NDIM * FDIM);
  #pragma unroll
  for (int j = 0; j < 4; ++j) {
    const int col = (ntB + j) * 16 + l;
    const float bias = b2[col];
    float partial = 0.0f;
    #pragma unroll
    for (int i = 0; i < 8; ++i) {
      const int row  = m0 + hi * 8 + i;                 // neighbor slot k
      const float ft = acc2[j][i] + bias;               // filter[row][col]
      const float fv = featB[(size_t)sNbr[row] * FDIM + col];
      partial += ft * sMask[row] * fv;
    }
    atomicAdd(&sOut[col], partial);  // ds_add_f32; 8 adders per column
  }
  __syncthreads();

  if (tid < FDIM) out[(size_t)bn * FDIM + tid] = sOut[tid];
}

extern "C" void kernel_launch(void* const* d_in, const int* in_sizes, int n_in,
                              void* d_out, int out_size, void* d_ws, size_t ws_size,
                              hipStream_t stream) {
  const float* features = (const float*)d_in[0];   // [32,512,128]
  const float* rbf      = (const float*)d_in[1];   // [32,512,64,64]
  const int*   nbr      = (const int*)d_in[2];     // [32,512,64]
  const float* mask     = (const float*)d_in[3];   // [32,512,64]
  const float* W1       = (const float*)d_in[4];   // [64,128]
  const float* b1       = (const float*)d_in[5];   // [128]
  const float* W2       = (const float*)d_in[6];   // [128,128]
  const float* b2       = (const float*)d_in[7];   // [128]
  float* out = (float*)d_out;                      // [32,512,128]

  const int w1Elems = 2 * 8 * 32 * 16;   // 8192 bf16
  const int w2Elems = 4 * 8 * 32 * 16;   // 16384 bf16
  __bf16* w1p = (__bf16*)d_ws;
  __bf16* w2p = (__bf16*)((char*)d_ws + (size_t)w1Elems * sizeof(__bf16));

  pack_w_bf16<<<w1Elems / 256, 256, 0, stream>>>(W1, w1p, 2);
  pack_w_bf16<<<w2Elems / 256, 256, 0, stream>>>(W2, w2p, 4);
  cfconv_fused<<<BDIM * NDIM, 256, 0, stream>>>(features, rbf, nbr, mask,
                                                b1, b2, w1p, w2p, out);
}
